// Attention_59090160058985
// MI455X (gfx1250) — matrix operand order repaired and checked by reading, not compiled, not run on hardware
//
#include <hip/hip_runtime.h>
#include <hip/hip_bf16.h>

typedef __attribute__((ext_vector_type(16))) __bf16 v16bf;
typedef __attribute__((ext_vector_type(8)))  __bf16 v8bf;
typedef __attribute__((ext_vector_type(8)))  float  v8f;

#define WMMA_BF16(a,b,c) __builtin_amdgcn_wmma_f32_16x16x32_bf16(false,(a),false,(b),(short)0,(c),false,false)

static constexpr int cB = 4, cS = 2048, cD = 768, cH = 24;
static constexpr int cNOPE = 32, cROPE = 16, cVD = 32, cQHD = 48, cQR = 384, cKVR = 128;
static constexpr int cM = cB * cS;            // 8192 token rows
static constexpr int cQW = cH * cQHD;         // 1152
static constexpr int cKVW = cH * (cNOPE+cVD); // 1536
static constexpr int cCKVW = cKVR + cROPE;    // 144
static constexpr int cPAD = 64;               // qs/ks padded head dim

// A-fragment loader: 16-bit A-matrix 16x32 layout (two contiguous 16B chunks per lane)
__device__ __forceinline__ v16bf load_afrag(const __bf16* p, int half) {
    v8bf l0 = *(const v8bf*)(p + 8 * half);
    v8bf l1 = *(const v8bf*)(p + 16 + 8 * half);
    v16bf a;
    #pragma unroll
    for (int i = 0; i < 8; ++i) { a[i] = l0[i]; a[8 + i] = l1[i]; }
    return a;
}

// ---------------------------------------------------------------- converts
__global__ void cvt_f32_bf16(const float* __restrict__ in, __bf16* __restrict__ out, int n) {
    for (int i = blockIdx.x * blockDim.x + threadIdx.x; i < n; i += gridDim.x * blockDim.x)
        out[i] = (__bf16)in[i];
}

// --------------------------------------------------- generic bf16 WMMA GEMM
// C[M,N](f32) = A[M,K](bf16,row-major) * B[K,N](bf16,row-major)
// Per-wave tile: 32 x (16*NT).  grid = (N/(16*NT), M/128), block = 128 (4 waves).
template<int NT>
__global__ void __launch_bounds__(128, 1)
gemm_bf16_wmma(const __bf16* __restrict__ A, const __bf16* __restrict__ B,
               float* __restrict__ C, int K, int lda, int ldb, int ldc) {
    const int lane = threadIdx.x & 31;
    const int wave = threadIdx.x >> 5;
    const int half = lane >> 4;
    const int lidx = lane & 15;
    const int m0 = (blockIdx.y * 4 + wave) * 32;
    const int n0 = blockIdx.x * (16 * NT);
    const __bf16* ap0 = A + (size_t)(m0 + lidx) * lda;
    const __bf16* ap1 = A + (size_t)(m0 + 16 + lidx) * lda;
    v8f acc[2][NT];
    #pragma unroll
    for (int mt = 0; mt < 2; ++mt)
        #pragma unroll
        for (int nt = 0; nt < NT; ++nt) acc[mt][nt] = v8f{};
    for (int k0 = 0; k0 < K; k0 += 32) {
        v16bf a0 = load_afrag(ap0 + k0, half);
        v16bf a1 = load_afrag(ap1 + k0, half);
        const __bf16* bp = B + (size_t)k0 * ldb + n0 + lidx;
        #pragma unroll
        for (int nt = 0; nt < NT; ++nt) {
            v16bf b;
            #pragma unroll
            for (int i = 0; i < 8; ++i) {
                b[i]     = bp[(size_t)(8 * half + i) * ldb + nt * 16];
                b[8 + i] = bp[(size_t)(16 + 8 * half + i) * ldb + nt * 16];
            }
            acc[0][nt] = WMMA_BF16(a0, b, acc[0][nt]);
            acc[1][nt] = WMMA_BF16(a1, b, acc[1][nt]);
        }
    }
    #pragma unroll
    for (int mt = 0; mt < 2; ++mt)
        #pragma unroll
        for (int nt = 0; nt < NT; ++nt)
            #pragma unroll
            for (int r = 0; r < 8; ++r)
                C[(size_t)(m0 + mt * 16 + r + 8 * half) * ldc + n0 + nt * 16 + lidx] = acc[mt][nt][r];
}

// ------------------------------------------------------------- RMS norms
__global__ void rmsnorm_bf16(const float* __restrict__ in, int in_stride, int width,
                             const float* __restrict__ gamma,
                             __bf16* __restrict__ out, int out_stride) {
    __shared__ float red[128];
    const int row = blockIdx.x;
    const float* x = in + (size_t)row * in_stride;
    float s = 0.f;
    for (int c = threadIdx.x; c < width; c += blockDim.x) { float v = x[c]; s += v * v; }
    red[threadIdx.x] = s; __syncthreads();
    for (int off = 64; off > 0; off >>= 1) {
        if (threadIdx.x < off) red[threadIdx.x] += red[threadIdx.x + off];
        __syncthreads();
    }
    const float r = rsqrtf(red[0] / (float)width + 1e-5f);
    __bf16* o = out + (size_t)row * out_stride;
    for (int c = threadIdx.x; c < width; c += blockDim.x)
        o[c] = (__bf16)(x[c] * r * gamma[c]);
}

// ------------------------------------------------- assemble qs / ks^T / v
// NOTE: reference swaps the rope parts: q gets rot(k_pe), k gets rot(q_pe).
__global__ void build_qkv(const float* __restrict__ q, const float* __restrict__ kv,
                          const float* __restrict__ ckv, const float* __restrict__ fcos,
                          const float* __restrict__ fsin,
                          __bf16* __restrict__ qsb,   // [B*H][S][64]  (48 used, 16 zero pad)
                          __bf16* __restrict__ kst,   // [B*H][64][S]  dim-major (transposed)
                          __bf16* __restrict__ vbuf)  // [B*H][S][32]
{
    const int t = blockIdx.x * blockDim.x + threadIdx.x;
    if (t >= cM * cH) return;
    const int h = t % cH, bs = t / cH;
    const int b = bs / cS, s = bs % cS;
    const int bh = b * cH + h;
    const float* qrow  = q   + (size_t)bs * cQW  + h * cQHD;
    const float* kvrow = kv  + (size_t)bs * cKVW + h * (cNOPE + cVD);
    const float* kpe   = ckv + (size_t)bs * cCKVW + cKVR;
    const float* cs = fcos + s * (cROPE / 2);
    const float* sn = fsin + s * (cROPE / 2);

    __bf16* qd = qsb  + ((size_t)bh * cS + s) * cPAD;
    __bf16* vd = vbuf + ((size_t)bh * cS + s) * cVD;
    __bf16* kd = kst  + (size_t)bh * cPAD * cS + s;   // dim d lives at kd[d*cS]

    #pragma unroll
    for (int i = 0; i < cNOPE; ++i) qd[i] = (__bf16)qrow[i];
    #pragma unroll
    for (int i = 0; i < cROPE / 2; ++i) {              // q rope part = rot(k_pe)
        float c = cs[i], si = sn[i];
        float xr = kpe[2 * i], xi = kpe[2 * i + 1];
        qd[cNOPE + 2 * i]     = (__bf16)(xr * c - xi * si);
        qd[cNOPE + 2 * i + 1] = (__bf16)(xr * si + xi * c);
    }
    #pragma unroll
    for (int i = cQHD; i < cPAD; ++i) qd[i] = (__bf16)0.f;

    #pragma unroll
    for (int i = 0; i < cNOPE; ++i) kd[(size_t)i * cS] = (__bf16)kvrow[i];
    const float* qpe = qrow + cNOPE;
    #pragma unroll
    for (int i = 0; i < cROPE / 2; ++i) {              // k rope part = rot(q_pe)
        float c = cs[i], si = sn[i];
        float xr = qpe[2 * i], xi = qpe[2 * i + 1];
        kd[(size_t)(cNOPE + 2 * i) * cS]     = (__bf16)(xr * c - xi * si);
        kd[(size_t)(cNOPE + 2 * i + 1) * cS] = (__bf16)(xr * si + xi * c);
    }
    #pragma unroll
    for (int i = cQHD; i < cPAD; ++i) kd[(size_t)i * cS] = (__bf16)0.f;

    #pragma unroll
    for (int i = 0; i < cVD; ++i) vd[i] = (__bf16)kvrow[cNOPE + i];
}

// -------------------------------------------------------- flash attention
// One wave per (bh, 32-query tile); 32-key tiles; online softmax in f32,
// one row per lane (all 32 lanes).  Single-wave workgroup => __syncthreads
// lowers to waits only (ISA 3.1: single-wave barrier ops become S_NOP).
// launch_bounds(32,1): let the allocator keep Q frags + accumulators resident
// (round-3 asm showed scratch_load_b128 spills of the Q fragments).
__global__ void __launch_bounds__(32, 1)
mla_attention(const __bf16* __restrict__ qsb, const __bf16* __restrict__ kst,
              const __bf16* __restrict__ vbuf, __bf16* __restrict__ aob) {
    __shared__ float  sS[32 * 32];
    __shared__ __bf16 sP[32 * 32];
    __shared__ float  sA[32];
    __shared__ float  sL[32];
    const int lane = threadIdx.x;
    const int half = lane >> 4;
    const int lidx = lane & 15;
    const int m0 = blockIdx.x * 32;
    const int bh = blockIdx.y;
    const float scale = 0.17677669529663687f;   // 1/sqrt(32)

    // Q fragments: 2 M-tiles x 2 K-chunks (head dim padded to 64)
    v16bf aq[2][2];
    #pragma unroll
    for (int mt = 0; mt < 2; ++mt) {
        const __bf16* qp = qsb + ((size_t)bh * cS + m0 + mt * 16 + lidx) * cPAD;
        aq[mt][0] = load_afrag(qp, half);
        aq[mt][1] = load_afrag(qp + 32, half);
    }
    v8f acc[2][2];
    #pragma unroll
    for (int mt = 0; mt < 2; ++mt) { acc[mt][0] = v8f{}; acc[mt][1] = v8f{}; }
    float m_i = -1e30f, l_i = 0.f;
    const __bf16* kbase = kst  + (size_t)bh * cPAD * cS;
    const __bf16* vbase = vbuf + (size_t)bh * cS * cVD;
    const int qglob = m0 + lane;                // softmax row for this lane

    for (int kt = 0; kt <= m0 + 31; kt += 32) {
        // ---- scores: 32x32 via 8 wmma (K frags reused across both M-tiles)
        v8f sc[2][2];
        #pragma unroll
        for (int mt = 0; mt < 2; ++mt) { sc[mt][0] = v8f{}; sc[mt][1] = v8f{}; }
        #pragma unroll
        for (int kc = 0; kc < 2; ++kc) {
            const __bf16* kp = kbase + (size_t)(kc * 32) * cS + kt + lidx;
            v16bf b0, b1;
            #pragma unroll
            for (int i = 0; i < 8; ++i) {
                const size_t r0 = (size_t)(8 * half + i) * cS, r1 = (size_t)(16 + 8 * half + i) * cS;
                b0[i] = kp[r0]; b0[8 + i] = kp[r1];
                b1[i] = kp[r0 + 16]; b1[8 + i] = kp[r1 + 16];
            }
            #pragma unroll
            for (int mt = 0; mt < 2; ++mt) {
                sc[mt][0] = WMMA_BF16(aq[mt][kc], b0, sc[mt][0]);
                sc[mt][1] = WMMA_BF16(aq[mt][kc], b1, sc[mt][1]);
            }
        }
        #pragma unroll
        for (int mt = 0; mt < 2; ++mt)
            #pragma unroll
            for (int nt = 0; nt < 2; ++nt)
                #pragma unroll
                for (int r = 0; r < 8; ++r)
                    sS[(mt * 16 + r + 8 * half) * 32 + nt * 16 + lidx] = sc[mt][nt][r];
        __syncthreads();

        // ---- online softmax: one row per lane, two LDS passes (max, then exp)
        {
            float tmax = -1e30f;
            #pragma unroll
            for (int c = 0; c < 32; ++c) {
                float v = (kt + c <= qglob) ? sS[lane * 32 + c] * scale : -1e30f;
                tmax = fmaxf(tmax, v);
            }
            const float mnew  = fmaxf(m_i, tmax);
            const float alpha = __expf(m_i - mnew);
            float psum = 0.f;
            #pragma unroll
            for (int c = 0; c < 32; ++c) {
                float p = (kt + c <= qglob) ? __expf(sS[lane * 32 + c] * scale - mnew) : 0.f;
                sP[lane * 32 + c] = (__bf16)p;
                psum += p;
            }
            l_i = l_i * alpha + psum;
            m_i = mnew;
            sA[lane] = alpha;
        }
        __syncthreads();

        // ---- rescale accumulators, then P @ V (4 wmma, V frags shared)
        #pragma unroll
        for (int mt = 0; mt < 2; ++mt)
            #pragma unroll
            for (int r = 0; r < 8; ++r) {
                const float al = sA[mt * 16 + r + 8 * half];
                acc[mt][0][r] *= al; acc[mt][1][r] *= al;
            }
        const __bf16* vp = vbase + (size_t)kt * cVD + lidx;
        v16bf bv0, bv1;
        #pragma unroll
        for (int i = 0; i < 8; ++i) {
            const size_t r0 = (size_t)(8 * half + i) * cVD, r1 = (size_t)(16 + 8 * half + i) * cVD;
            bv0[i] = vp[r0]; bv0[8 + i] = vp[r1];
            bv1[i] = vp[r0 + 16]; bv1[8 + i] = vp[r1 + 16];
        }
        #pragma unroll
        for (int mt = 0; mt < 2; ++mt) {
            const __bf16* pp = sP + (mt * 16 + lidx) * 32;
            v8bf c0 = *(const v8bf*)(pp + 8 * half);
            v8bf c1 = *(const v8bf*)(pp + 16 + 8 * half);
            v16bf ap;
            #pragma unroll
            for (int i = 0; i < 8; ++i) { ap[i] = c0[i]; ap[8 + i] = c1[i]; }
            acc[mt][0] = WMMA_BF16(ap, bv0, acc[mt][0]);
            acc[mt][1] = WMMA_BF16(ap, bv1, acc[mt][1]);
        }
        __syncthreads();
    }
    sL[lane] = l_i;
    __syncthreads();
    const int b = bh / cH, h = bh % cH;
    #pragma unroll
    for (int mt = 0; mt < 2; ++mt)
        #pragma unroll
        for (int r = 0; r < 8; ++r) {
            const int srow = m0 + mt * 16 + r + 8 * half;
            const float inv = 1.f / sL[mt * 16 + r + 8 * half];
            const size_t base = ((size_t)(b * cS + srow)) * (cH * cVD) + h * cVD;
            aob[base + lidx]      = (__bf16)(acc[mt][0][r] * inv);
            aob[base + 16 + lidx] = (__bf16)(acc[mt][1][r] * inv);
        }
}

// ----------------------------------------------------------------- launch
extern "C" void kernel_launch(void* const* d_in, const int* in_sizes, int n_in,
                              void* d_out, int out_size, void* d_ws, size_t ws_size,
                              hipStream_t stream) {
    const float* x     = (const float*)d_in[0];
    // d_in[1] = mask (unused: causal mask applied analytically)
    const float* fcos  = (const float*)d_in[2];
    const float* fsin  = (const float*)d_in[3];
    const float* Wqa   = (const float*)d_in[4];
    const float* qa_ln = (const float*)d_in[5];
    const float* Wqb   = (const float*)d_in[6];
    const float* Wkva  = (const float*)d_in[7];
    const float* kv_ln = (const float*)d_in[8];
    const float* Wkvb  = (const float*)d_in[9];
    const float* Wo    = (const float*)d_in[10];
    float* out = (float*)d_out;

    char* ws = (char*)d_ws;
    size_t off = 0;
    auto alloc = [&](size_t bytes) -> void* {
        void* p = ws + off;
        off = (off + bytes + 255) & ~(size_t)255;
        return p;
    };
    __bf16* xb    = (__bf16*)alloc((size_t)cM * cD * 2);
    __bf16* wqab  = (__bf16*)alloc((size_t)cD * cQR * 2);
    __bf16* wqbb  = (__bf16*)alloc((size_t)cQR * cQW * 2);
    __bf16* wkvab = (__bf16*)alloc((size_t)cD * cCKVW * 2);
    __bf16* wkvbb = (__bf16*)alloc((size_t)cKVR * cKVW * 2);
    __bf16* wob   = (__bf16*)alloc((size_t)(cH * cVD) * cD * 2);
    float*  qa    = (float*) alloc((size_t)cM * cQR * 4);
    float*  ckv   = (float*) alloc((size_t)cM * cCKVW * 4);
    __bf16* qan   = (__bf16*)alloc((size_t)cM * cQR * 2);
    __bf16* ckvn  = (__bf16*)alloc((size_t)cM * cKVR * 2);
    float*  qf    = (float*) alloc((size_t)cM * cQW * 4);
    float*  kvf   = (float*) alloc((size_t)cM * cKVW * 4);
    __bf16* qsb   = (__bf16*)alloc((size_t)cB * cH * cS * cPAD * 2);
    __bf16* kst   = (__bf16*)alloc((size_t)cB * cH * cPAD * cS * 2);
    __bf16* vbuf  = (__bf16*)alloc((size_t)cB * cH * cS * cVD * 2);
    __bf16* aob   = (__bf16*)alloc((size_t)cM * (cH * cVD) * 2);

    auto cvt = [&](const float* src, __bf16* dst, int n) {
        cvt_f32_bf16<<<(n + 255) / 256, 256, 0, stream>>>(src, dst, n);
    };
    cvt(x,    xb,    cM * cD);
    cvt(Wqa,  wqab,  cD * cQR);
    cvt(Wqb,  wqbb,  cQR * cQW);
    cvt(Wkva, wkvab, cD * cCKVW);
    cvt(Wkvb, wkvbb, cKVR * cKVW);
    cvt(Wo,   wob,   cH * cVD * cD);

    // qa = x @ Wqa ; ckv = x @ Wkva   (N=144 not /64 -> NT=1 variant)
    gemm_bf16_wmma<4><<<dim3(cQR / 64,   cM / 128), 128, 0, stream>>>(xb, wqab,  qa,  cD, cD, cQR,  cQR);
    gemm_bf16_wmma<1><<<dim3(cCKVW / 16, cM / 128), 128, 0, stream>>>(xb, wkvab, ckv, cD, cD, cCKVW, cCKVW);

    rmsnorm_bf16<<<cM, 128, 0, stream>>>(qa,  cQR,   cQR,  qa_ln, qan,  cQR);
    rmsnorm_bf16<<<cM, 128, 0, stream>>>(ckv, cCKVW, cKVR, kv_ln, ckvn, cKVR);

    // q = qa_n @ Wqb ; kv = ckv_n @ Wkvb
    gemm_bf16_wmma<4><<<dim3(cQW / 64,  cM / 128), 128, 0, stream>>>(qan,  wqbb,  qf,  cQR,  cQR,  cQW,  cQW);
    gemm_bf16_wmma<4><<<dim3(cKVW / 64, cM / 128), 128, 0, stream>>>(ckvn, wkvbb, kvf, cKVR, cKVR, cKVW, cKVW);

    build_qkv<<<(cM * cH + 255) / 256, 256, 0, stream>>>(qf, kvf, ckv, fcos, fsin, qsb, kst, vbuf);

    mla_attention<<<dim3(cS / 32, cB * cH), 32, 0, stream>>>(qsb, kst, vbuf, aob);

    // out = attn_out @ Wo
    gemm_bf16_wmma<4><<<dim3(cD / 64, cM / 128), 128, 0, stream>>>(aob, wob, out, cH * cVD, cH * cVD, cD, cD);
}
